// BaselineGAT128_87256555585812
// MI455X (gfx1250) — compile-verified, matrix operand order checked by reading, not run on hardware
//
#include <hip/hip_runtime.h>
#include <hip/hip_bf16.h>
#include <stdint.h>

// ---------------------------------------------------------------------------
// Problem constants (from reference)
// ---------------------------------------------------------------------------
#define NN       100000
#define EE       1600000
#define NH       4
#define NC       128
#define DD       512
#define FIN      182
#define KP1      192                 // FIN padded to multiple of 32
#define MPAD     100096              // NN padded to multiple of 128
#define NEGSLOPE 0.2f
#define LNEPS    1e-5f

typedef __attribute__((ext_vector_type(16))) __bf16 v16bf;
typedef __attribute__((ext_vector_type(8)))  __bf16 v8bf;
typedef __attribute__((ext_vector_type(8)))  float  v8f;

// ---------------------------------------------------------------------------
// CDNA5 async global->LDS copy (tracked by ASYNCcnt) + wait helper
// ---------------------------------------------------------------------------
__device__ __forceinline__ void async_b128_to_lds(const void* gptr, void* lptr) {
    unsigned lds_off = (unsigned)(uintptr_t)lptr;          // addr[31:0] = LDS offset
    unsigned long long ga = (unsigned long long)(uintptr_t)gptr;
    asm volatile("global_load_async_to_lds_b128 %0, %1, off"
                 :: "v"(lds_off), "v"(ga) : "memory");
}
__device__ __forceinline__ void wait_async0() {
    asm volatile("s_wait_asynccnt 0x0" ::: "memory");
}

// ---------------------------------------------------------------------------
// fp32 -> bf16 conversions with zero padding
// ---------------------------------------------------------------------------
__global__ void k_conv_x(const float* __restrict__ x, __bf16* __restrict__ xb) {
    int i = blockIdx.x * blockDim.x + threadIdx.x;
    int total = MPAD * KP1;
    if (i >= total) return;
    int row = i / KP1, col = i - row * KP1;
    float v = (row < NN && col < FIN) ? x[(size_t)row * FIN + col] : 0.0f;
    xb[i] = (__bf16)v;
}

// W[K,N] fp32 -> Wt[N,Kpad] bf16 (transposed + K zero-padded): makes the GEMM
// B-tile a contiguous row-major load (no LDS transpose scatter needed).
__global__ void k_conv_wt(const float* __restrict__ w, __bf16* __restrict__ wt,
                          int ksrc, int kpad, int nout) {
    int i = blockIdx.x * blockDim.x + threadIdx.x;
    int total = nout * kpad;
    if (i >= total) return;
    int n = i / kpad, k = i - n * kpad;
    float v = (k < ksrc) ? w[(size_t)k * nout + n] : 0.0f;
    wt[i] = (__bf16)v;
}

// ---------------------------------------------------------------------------
// bf16 WMMA GEMM: C[M,Nw] = A[M,K] @ Bt[Nw,K]^T (+bias, +relu)
// block = 256 threads = 8 waves; tile 128(M) x 64(N); waves in 4x2 grid,
// each wave computes 32x32 via 2x2 v_wmma_f32_16x16x32_bf16 fragments.
// Tiles are streamed into double-buffered LDS with GLOBAL_LOAD_ASYNC_TO_LDS,
// software-pipelined against the WMMA compute.
// M multiple of 128, K multiple of 32, Nw multiple of 64 (no guards needed).
// ---------------------------------------------------------------------------
#define BM 128
#define BN 64
#define BK 32

__global__ __launch_bounds__(256) void k_gemm_bf16(
        const __bf16* __restrict__ A, const __bf16* __restrict__ Bt,
        float* __restrict__ C, int K, int Nw,
        const float* __restrict__ bias, int do_relu) {
    __shared__ __align__(16) __bf16 As[2][BM * BK];
    __shared__ __align__(16) __bf16 Bs[2][BN * BK];   // Bs[n*BK + k]

    const int tid   = threadIdx.x;
    const int wave  = tid >> 5;
    const int lane  = tid & 31;
    const int wm    = wave >> 1;        // 0..3
    const int wn    = wave & 1;         // 0..1
    const int lrow  = lane & 15;
    const int lhalf = lane >> 4;
    const int bm    = blockIdx.y * BM;
    const int bn    = blockIdx.x * BN;

    // per-thread tile-load coordinates (8-bf16 = 16B vectors)
    const int arow0 = tid >> 2,            acq0 = tid & 3;          // A vec 0
    const int arow1 = (tid + 256) >> 2,    acq1 = tid & 3;          // A vec 1
    const int brow  = tid >> 2,            bcq  = tid & 3;          // B vec

    v8f acc[2][2];
#pragma unroll
    for (int mi = 0; mi < 2; ++mi)
#pragma unroll
        for (int ni = 0; ni < 2; ++ni)
#pragma unroll
            for (int r = 0; r < 8; ++r) acc[mi][ni][r] = 0.0f;

    auto issue_tile = [&](int k0, int b) {
        async_b128_to_lds(A  + (size_t)(bm + arow0) * K + k0 + acq0 * 8,
                          &As[b][arow0 * BK + acq0 * 8]);
        async_b128_to_lds(A  + (size_t)(bm + arow1) * K + k0 + acq1 * 8,
                          &As[b][arow1 * BK + acq1 * 8]);
        async_b128_to_lds(Bt + (size_t)(bn + brow)  * K + k0 + bcq  * 8,
                          &Bs[b][brow * BK + bcq * 8]);
    };

    const int nk = K / BK;
    issue_tile(0, 0);
    wait_async0();
    __syncthreads();

    for (int kk = 0; kk < nk; ++kk) {
        const int cur = kk & 1;
        if (kk + 1 < nk) issue_tile((kk + 1) * BK, cur ^ 1);

        v16bf afr[2], bfr[2];
#pragma unroll
        for (int mi = 0; mi < 2; ++mi) {
            int r = wm * 32 + mi * 16 + lrow;
            v8bf* ap = (v8bf*)&afr[mi];
            ap[0] = *(v8bf*)(&As[cur][r * BK + lhalf * 8]);       // K = 8*half+0..7
            ap[1] = *(v8bf*)(&As[cur][r * BK + 16 + lhalf * 8]);  // K = 16+8*half+..
        }
#pragma unroll
        for (int ni = 0; ni < 2; ++ni) {
            int c = wn * 32 + ni * 16 + lrow;
            v8bf* bp = (v8bf*)&bfr[ni];
            bp[0] = *(v8bf*)(&Bs[cur][c * BK + lhalf * 8]);
            bp[1] = *(v8bf*)(&Bs[cur][c * BK + 16 + lhalf * 8]);
        }
#pragma unroll
        for (int mi = 0; mi < 2; ++mi)
#pragma unroll
            for (int ni = 0; ni < 2; ++ni)
                acc[mi][ni] = __builtin_amdgcn_wmma_f32_16x16x32_bf16(
                    false, afr[mi], false, bfr[ni], (short)0, acc[mi][ni],
                    false, false);

        if (kk + 1 < nk) {
            wait_async0();      // next tile resident in LDS
            __syncthreads();    // + everyone done reading current tile
        }
    }

    // ---- epilogue: C layout M = r + 8*half, N = lane&15
#pragma unroll
    for (int mi = 0; mi < 2; ++mi) {
#pragma unroll
        for (int ni = 0; ni < 2; ++ni) {
            int col = bn + wn * 32 + ni * 16 + lrow;
            float bv = bias ? bias[col] : 0.0f;
#pragma unroll
            for (int r = 0; r < 8; ++r) {
                int rowg = bm + wm * 32 + mi * 16 + r + 8 * lhalf;
                float v = acc[mi][ni][r] + bv;
                if (do_relu) v = fmaxf(v, 0.0f);
                C[(size_t)rowg * Nw + col] = v;
            }
        }
    }
}

// ---------------------------------------------------------------------------
// Per-node attention score dots: s[n,h] = sum_c h[n,h,c] * a[h,c]
// ---------------------------------------------------------------------------
__global__ void k_scores(const float* __restrict__ h,
                         const float* __restrict__ a_src,
                         const float* __restrict__ a_dst,
                         float* __restrict__ ssrc, float* __restrict__ sdst) {
    int i = blockIdx.x * blockDim.x + threadIdx.x;
    if (i >= NN * NH) return;
    int n = i >> 2, hh = i & 3;
    const float4* hp = (const float4*)(h + (size_t)n * DD + hh * NC);
    const float4* ap = (const float4*)(a_src + hh * NC);
    const float4* dp = (const float4*)(a_dst + hh * NC);
    float s0 = 0.f, s1 = 0.f;
#pragma unroll 8
    for (int k = 0; k < NC / 4; ++k) {
        float4 v = hp[k];
        float4 a = ap[k];
        float4 d = dp[k];
        s0 += v.x * a.x + v.y * a.y + v.z * a.z + v.w * a.w;
        s1 += v.x * d.x + v.y * d.y + v.z * d.z + v.w * d.w;
    }
    ssrc[i] = s0;
    sdst[i] = s1;
}

// ---------------------------------------------------------------------------
// Segment softmax (ordered-int float max trick)
// ---------------------------------------------------------------------------
__device__ __forceinline__ int fkey(float f) {
    int i = __float_as_int(f);
    return i >= 0 ? i : (i ^ 0x7fffffff);
}
__device__ __forceinline__ float fdecode(int k) {
    int i = k >= 0 ? k : (k ^ 0x7fffffff);
    return __int_as_float(i);
}

__global__ void k_init_edge(int* __restrict__ emaxI, float* __restrict__ denom) {
    int i = blockIdx.x * blockDim.x + threadIdx.x;
    if (i >= NN * NH) return;
    emaxI[i] = 0x80000000;   // -inf key (every node has a self-loop -> updated)
    denom[i] = 0.0f;
}

__device__ __forceinline__ void edge_sd(const long long* ei, int e,
                                        int& src, int& dst) {
    if (e < EE) {
        src = (int)ei[e];
        dst = (int)ei[EE + e];
    } else {
        src = dst = e - EE;   // self loop
    }
}

__global__ void k_edge_max(const long long* __restrict__ ei,
                           const float* __restrict__ ssrc,
                           const float* __restrict__ sdst,
                           int* __restrict__ emaxI) {
    int idx = blockIdx.x * blockDim.x + threadIdx.x;
    if (idx >= (EE + NN) * NH) return;
    int e = idx >> 2, hh = idx & 3;
    int src, dst;
    edge_sd(ei, e, src, dst);
    float sc = ssrc[src * NH + hh] + sdst[dst * NH + hh];
    sc = sc > 0.0f ? sc : NEGSLOPE * sc;
    atomicMax(&emaxI[dst * NH + hh], fkey(sc));
}

__global__ void k_edge_exp(const long long* __restrict__ ei,
                           const float* __restrict__ ssrc,
                           const float* __restrict__ sdst,
                           const int* __restrict__ emaxI,
                           float* __restrict__ denom,
                           float* __restrict__ pbuf) {
    int idx = blockIdx.x * blockDim.x + threadIdx.x;
    if (idx >= (EE + NN) * NH) return;
    int e = idx >> 2, hh = idx & 3;
    int src, dst;
    edge_sd(ei, e, src, dst);
    float sc = ssrc[src * NH + hh] + sdst[dst * NH + hh];
    sc = sc > 0.0f ? sc : NEGSLOPE * sc;
    float m = fdecode(emaxI[dst * NH + hh]);
    float pv = __expf(sc - m);
    pbuf[idx] = pv;
    atomicAdd(&denom[dst * NH + hh], pv);
}

__global__ void k_init_agg(float* __restrict__ agg, const float* __restrict__ b) {
    int i = blockIdx.x * blockDim.x + threadIdx.x;
    if (i >= NN * DD) return;
    agg[i] = b[i & (DD - 1)];
}

// one 128-thread block per edge; each thread moves 4 contiguous floats
__global__ __launch_bounds__(128) void k_aggregate(
        const long long* __restrict__ ei, const float* __restrict__ pbuf,
        const float* __restrict__ denom, const float* __restrict__ hfeat,
        float* __restrict__ agg) {
    int e = blockIdx.x;
    int t = threadIdx.x;
    int src, dst;
    edge_sd(ei, e, src, dst);
    int hh = t >> 5;                       // 32 threads per head (128 floats)
    float alpha = pbuf[e * NH + hh] / denom[dst * NH + hh];
    float4 v = *(const float4*)(hfeat + (size_t)src * DD + t * 4);
    float* o = agg + (size_t)dst * DD + t * 4;
    atomicAdd(o + 0, alpha * v.x);
    atomicAdd(o + 1, alpha * v.y);
    atomicAdd(o + 2, alpha * v.z);
    atomicAdd(o + 3, alpha * v.w);
}

// ---------------------------------------------------------------------------
// LayerNorm + ELU fused with bf16 down-convert (writes zero into M-pad rows)
// ---------------------------------------------------------------------------
__global__ __launch_bounds__(128) void k_ln_elu(
        const float* __restrict__ xin, const float* __restrict__ g,
        const float* __restrict__ be, __bf16* __restrict__ outb) {
    int row = blockIdx.x;
    int tid = threadIdx.x;
    size_t base = (size_t)row * DD + tid * 4;
    if (row >= NN) {
        outb[base + 0] = (__bf16)0.0f;
        outb[base + 1] = (__bf16)0.0f;
        outb[base + 2] = (__bf16)0.0f;
        outb[base + 3] = (__bf16)0.0f;
        return;
    }
    float4 v = *(const float4*)(xin + (size_t)row * DD + tid * 4);
    float s = v.x + v.y + v.z + v.w;
    float q = v.x * v.x + v.y * v.y + v.z * v.z + v.w * v.w;
#pragma unroll
    for (int off = 16; off > 0; off >>= 1) {
        s += __shfl_xor(s, off, 32);
        q += __shfl_xor(q, off, 32);
    }
    __shared__ float rs[4], rq[4];
    int wid = tid >> 5, lane = tid & 31;
    if (lane == 0) { rs[wid] = s; rq[wid] = q; }
    __syncthreads();
    float S = rs[0] + rs[1] + rs[2] + rs[3];
    float Q = rq[0] + rq[1] + rq[2] + rq[3];
    float mu  = S * (1.0f / DD);
    float var = Q * (1.0f / DD) - mu * mu;
    float inv = rsqrtf(var + LNEPS);
    int d = tid * 4;
    float xs[4] = {v.x, v.y, v.z, v.w};
#pragma unroll
    for (int j = 0; j < 4; ++j) {
        float y = (xs[j] - mu) * inv * g[d + j] + be[d + j];
        y = y > 0.0f ? y : (__expf(y) - 1.0f);
        outb[base + j] = (__bf16)y;
    }
}

// ---------------------------------------------------------------------------
// Final skinny GEMM: out[n, 0:2] = z[n,:] @ Wc2[512,2] + bc2
// ---------------------------------------------------------------------------
__global__ void k_final(const float* __restrict__ z, const float* __restrict__ w,
                        const float* __restrict__ b, float* __restrict__ out) {
    int n = blockIdx.x * blockDim.x + threadIdx.x;
    if (n >= NN) return;
    const float4* zp = (const float4*)(z + (size_t)n * DD);
    float a0 = 0.f, a1 = 0.f;
#pragma unroll 4
    for (int k = 0; k < DD / 4; ++k) {
        float4 v = zp[k];
        const float* wr = w + k * 8;
        a0 += v.x * wr[0] + v.y * wr[2] + v.z * wr[4] + v.w * wr[6];
        a1 += v.x * wr[1] + v.y * wr[3] + v.z * wr[5] + v.w * wr[7];
    }
    out[n * 2 + 0] = a0 + b[0];
    out[n * 2 + 1] = a1 + b[1];
}

// ---------------------------------------------------------------------------
// Host-side launch sequence
// ---------------------------------------------------------------------------
static inline char* carve(char*& p, size_t bytes) {
    char* r = p;
    p += (bytes + 255) & ~(size_t)255;
    return r;
}

extern "C" void kernel_launch(void* const* d_in, const int* in_sizes, int n_in,
                              void* d_out, int out_size, void* d_ws, size_t ws_size,
                              hipStream_t stream) {
    (void)in_sizes; (void)n_in; (void)out_size; (void)ws_size;
    const float*     x      = (const float*)d_in[0];
    const long long* ei     = (const long long*)d_in[1];
    const float*     W1     = (const float*)d_in[2];
    const float*     a_src1 = (const float*)d_in[3];
    const float*     a_dst1 = (const float*)d_in[4];
    const float*     b1     = (const float*)d_in[5];
    const float*     g1     = (const float*)d_in[6];
    const float*     be1    = (const float*)d_in[7];
    const float*     W2     = (const float*)d_in[8];
    const float*     a_src2 = (const float*)d_in[9];
    const float*     a_dst2 = (const float*)d_in[10];
    const float*     b2     = (const float*)d_in[11];
    const float*     g2     = (const float*)d_in[12];
    const float*     be2    = (const float*)d_in[13];
    const float*     Wc1    = (const float*)d_in[14];
    const float*     bc1    = (const float*)d_in[15];
    const float*     Wc2    = (const float*)d_in[16];
    const float*     bc2    = (const float*)d_in[17];
    float*           out    = (float*)d_out;

    char* p = (char*)d_ws;
    __bf16* Xb   = (__bf16*)carve(p, (size_t)MPAD * KP1 * 2);
    __bf16* W1t  = (__bf16*)carve(p, (size_t)DD * KP1 * 2);   // [N=512][K=192]
    __bf16* W2t  = (__bf16*)carve(p, (size_t)DD * DD * 2);    // [512][512]
    __bf16* Wc1t = (__bf16*)carve(p, (size_t)DD * DD * 2);
    float*  bufA = (float*)carve(p, (size_t)MPAD * DD * 4);   // GEMM outputs
    float*  bufB = (float*)carve(p, (size_t)NN * DD * 4);     // aggregation
    __bf16* bufP = (__bf16*)carve(p, (size_t)MPAD * DD * 2);  // post LN/ELU bf16
    float*  ssrc = (float*)carve(p, (size_t)NN * NH * 4);
    float*  sdst = (float*)carve(p, (size_t)NN * NH * 4);
    int*    emaxI= (int*)carve(p, (size_t)NN * NH * 4);
    float*  denom= (float*)carve(p, (size_t)NN * NH * 4);
    float*  pbuf = (float*)carve(p, (size_t)(EE + NN) * NH * 4);

    const int TB = 256;
    auto blks = [](long long total, int tb) { return dim3((unsigned)((total + tb - 1) / tb)); };
    dim3 gemm_grid(DD / BN, MPAD / BM);

    // --- convert inputs to bf16 (A row-major K-padded; B transposed) ---
    k_conv_x<<<blks((long long)MPAD * KP1, TB), TB, 0, stream>>>(x, Xb);
    k_conv_wt<<<blks((long long)DD * KP1, TB), TB, 0, stream>>>(W1, W1t, FIN, KP1, DD);
    k_conv_wt<<<blks((long long)DD * DD, TB), TB, 0, stream>>>(W2, W2t, DD, DD, DD);
    k_conv_wt<<<blks((long long)DD * DD, TB), TB, 0, stream>>>(Wc1, Wc1t, DD, DD, DD);

    // ================= Layer 1 =================
    k_gemm_bf16<<<gemm_grid, 256, 0, stream>>>(Xb, W1t, bufA, KP1, DD, nullptr, 0);
    k_scores<<<blks((long long)NN * NH, TB), TB, 0, stream>>>(bufA, a_src1, a_dst1, ssrc, sdst);
    k_init_edge<<<blks((long long)NN * NH, TB), TB, 0, stream>>>(emaxI, denom);
    k_edge_max<<<blks((long long)(EE + NN) * NH, TB), TB, 0, stream>>>(ei, ssrc, sdst, emaxI);
    k_edge_exp<<<blks((long long)(EE + NN) * NH, TB), TB, 0, stream>>>(ei, ssrc, sdst, emaxI, denom, pbuf);
    k_init_agg<<<blks((long long)NN * DD, TB), TB, 0, stream>>>(bufB, b1);
    k_aggregate<<<dim3(EE + NN), 128, 0, stream>>>(ei, pbuf, denom, bufA, bufB);
    k_ln_elu<<<dim3(MPAD), 128, 0, stream>>>(bufB, g1, be1, bufP);

    // ================= Layer 2 =================
    k_gemm_bf16<<<gemm_grid, 256, 0, stream>>>(bufP, W2t, bufA, DD, DD, nullptr, 0);
    k_scores<<<blks((long long)NN * NH, TB), TB, 0, stream>>>(bufA, a_src2, a_dst2, ssrc, sdst);
    k_init_edge<<<blks((long long)NN * NH, TB), TB, 0, stream>>>(emaxI, denom);
    k_edge_max<<<blks((long long)(EE + NN) * NH, TB), TB, 0, stream>>>(ei, ssrc, sdst, emaxI);
    k_edge_exp<<<blks((long long)(EE + NN) * NH, TB), TB, 0, stream>>>(ei, ssrc, sdst, emaxI, denom, pbuf);
    k_init_agg<<<blks((long long)NN * DD, TB), TB, 0, stream>>>(bufB, b2);
    k_aggregate<<<dim3(EE + NN), 128, 0, stream>>>(ei, pbuf, denom, bufA, bufB);
    k_ln_elu<<<dim3(MPAD), 128, 0, stream>>>(bufB, g2, be2, bufP);

    // ================= Classifier =================
    k_gemm_bf16<<<gemm_grid, 256, 0, stream>>>(bufP, Wc1t, bufA, DD, DD, bc1, 1);
    k_final<<<blks(NN, TB), TB, 0, stream>>>(bufA, Wc2, bc2, out);
}
